// TanhAttention_85358180041263
// MI455X (gfx1250) — matrix operand order repaired and checked by reading, not compiled, not run on hardware
//
#include <hip/hip_runtime.h>
#include <hip/hip_bf16.h>

typedef __attribute__((ext_vector_type(16))) __bf16 v16bf;
typedef __attribute__((ext_vector_type(8)))  __bf16 v8bf;
typedef __attribute__((ext_vector_type(8)))  float  v8f;

#define NEG_INF (-1e30f)

__device__ __forceinline__ float fast_exp2(float x) {
#if __has_builtin(__builtin_amdgcn_exp2f)
    return __builtin_amdgcn_exp2f(x);
#else
    return exp2f(x);
#endif
}

__device__ __forceinline__ float fast_tanh(float x) {
#if __has_builtin(__builtin_amdgcn_tanhf)
    return __builtin_amdgcn_tanhf(x);           // v_tanh_f32 (CDNA5 TRANS op)
#elif __has_builtin(__builtin_amdgcn_tanh_f32)
    return __builtin_amdgcn_tanh_f32(x);
#else
    float e = fast_exp2(x * 2.885390082f);      // exp(2x)
    return 1.0f - 2.0f / (e + 1.0f);
#endif
}

__device__ __forceinline__ v8f wmma_bf16(v16bf a, v16bf b, v8f c) {
    // D = A(16x32 bf16) * B(32x16 bf16) + C(16x16 f32)
    return __builtin_amdgcn_wmma_f32_16x16x32_bf16(
        /*neg_a=*/false, a, /*neg_b=*/false, b,
        /*c_mod=*/(short)0, c, /*reuse_a=*/false, /*reuse_b=*/false);
}

// Async copy of 16 bytes global -> LDS (gfx1250 GLOBAL_LOAD_ASYNC_TO_LDS_B128,
// tracked by ASYNCcnt). Flat-pointer low 32 bits == LDS byte offset (ISA 10.2).
__device__ __forceinline__ void async_g2l_b128(const void* gptr, void* lptr) {
    unsigned lds = (unsigned)(unsigned long long)lptr;
    unsigned long long ga = (unsigned long long)gptr;
    asm volatile("global_load_async_to_lds_b128 %0, %1, off"
                 :: "v"(lds), "v"(ga) : "memory");
}
__device__ __forceinline__ void wait_asynccnt0() {
    asm volatile("s_wait_asynccnt 0x0" ::: "memory");
}

__device__ __forceinline__ v16bf make_afrag(v8bf lo, v8bf hi) {
    v16bf a;
    #pragma unroll
    for (int j = 0; j < 8; ++j) { a[j] = lo[j]; a[8 + j] = hi[j]; }
    return a;
}

// ---------------------------------------------------------------------------
// Kernel 0a: f32 -> bf16 convert
// ---------------------------------------------------------------------------
__global__ void ta_cvt_bf16(const float* __restrict__ x, __bf16* __restrict__ y, int n) {
    int i = blockIdx.x * 256 + threadIdx.x;
    if (i < n) y[i] = (__bf16)x[i];
}

// Kernel 0b: memory (B,Tm,256) f32 -> bf16 row-major AND bf16 transposed (B,256,Tm)
__global__ void ta_cvt_mem(const float* __restrict__ mem,
                           __bf16* __restrict__ mem_bf,
                           __bf16* __restrict__ memT) {
    int i = blockIdx.x * 256 + threadIdx.x;        // i < 4*1024*256
    float v = mem[i];
    mem_bf[i] = (__bf16)v;
    int d = i & 255;
    int m = (i >> 8) & 1023;
    int b = i >> 18;
    memT[((b << 8) + d) * 1024 + m] = (__bf16)v;
}

// ---------------------------------------------------------------------------
// Kernel 1: projection  C[M,64] = A[M,256](bf16) * W[64,256]^T(bf16) + bias
// One wave per 16x64 output strip (4 N-tiles); A-fragment reused by 4 WMMAs.
// 8 K-steps -> 32 wmma per wave.
// ---------------------------------------------------------------------------
__global__ void ta_proj(const __bf16* __restrict__ A, const __bf16* __restrict__ W,
                        const float* __restrict__ bias, float* __restrict__ C,
                        int Mtiles) {
    const int K = 256, N = 64;
    int mtile = blockIdx.x * 4 + (threadIdx.x >> 5);
    int lane  = threadIdx.x & 31;
    if (mtile >= Mtiles) return;

    int r  = lane & 15;
    int hi = lane >> 4;

    v8f c0, c1, c2, c3;
    {
        float b0 = bias ? bias[ 0 + r] : 0.0f;
        float b1 = bias ? bias[16 + r] : 0.0f;
        float b2 = bias ? bias[32 + r] : 0.0f;
        float b3 = bias ? bias[48 + r] : 0.0f;
        #pragma unroll
        for (int j = 0; j < 8; ++j) { c0[j] = b0; c1[j] = b1; c2[j] = b2; c3[j] = b3; }
    }

    const __bf16* arow = A + (mtile * 16 + r) * K;
    const __bf16* wr0 = W + ( 0 + r) * K;
    const __bf16* wr1 = W + (16 + r) * K;
    const __bf16* wr2 = W + (32 + r) * K;
    const __bf16* wr3 = W + (48 + r) * K;

    #pragma unroll
    for (int kb = 0; kb < 256; kb += 32) {
        v8bf alo = *(const v8bf*)(arow + kb + hi * 16);       // K = kb+hi*16 .. +7
        v8bf ahi = *(const v8bf*)(arow + kb + hi * 16 + 8);   // K = kb+hi*16+8 .. +7
        v16bf b0 = *(const v16bf*)(wr0 + kb + hi * 16);       // K = kb+hi*16 .. +15
        v16bf b1 = *(const v16bf*)(wr1 + kb + hi * 16);
        v16bf b2 = *(const v16bf*)(wr2 + kb + hi * 16);
        v16bf b3 = *(const v16bf*)(wr3 + kb + hi * 16);
        v16bf a = make_afrag(alo, ahi);
        c0 = wmma_bf16(a, b0, c0);
        c1 = wmma_bf16(a, b1, c1);
        c2 = wmma_bf16(a, b2, c2);
        c3 = wmma_bf16(a, b3, c3);
    }

    // C/D layout: lane -> rows hi*8+j, col r
    float* crow = C + (mtile * 16 + hi * 8) * N + r;
    #pragma unroll
    for (int j = 0; j < 8; ++j) {
        crow[j * N +  0] = c0[j];
        crow[j * N + 16] = c1[j];
        crow[j * N + 32] = c2[j];
        crow[j * N + 48] = c3[j];
    }
}

// ---------------------------------------------------------------------------
// Kernel 2: score[b,i,m] = sum_d Wf[d] * tanh(in_item[b,i,d] + mem_item[b,m,d])
// Block = 256 threads = 16(i) x 16(m) tile.  Tiles staged to LDS with
// GLOBAL_LOAD_ASYNC_TO_LDS_B128 (ASYNCcnt), stride 68 (16B-aligned rows,
// bank-conflict-free: stride mod 64 = 4).
// ---------------------------------------------------------------------------
__global__ void ta_score(const float* __restrict__ in_item,
                         const float* __restrict__ mem_item,
                         const float* __restrict__ W_final,
                         const int* __restrict__ ilen, const int* __restrict__ mlen,
                         float* __restrict__ score) {
    __shared__ float sIn [16 * 68];
    __shared__ float sMem[16 * 68];
    __shared__ float sWf [64];

    int tid = threadIdx.x;
    int b = blockIdx.z, it = blockIdx.y, mt = blockIdx.x;

    // Each thread async-copies one 16B chunk of each tile: row = tid>>4,
    // cols c4..c4+3 where c4 = (tid&15)*4.
    int row = tid >> 4;
    int c4  = (tid & 15) << 2;
    const float* gIn  = in_item  + ((b * 512  + it * 16 + row) << 6) + c4;
    const float* gMem = mem_item + ((b * 1024 + mt * 16 + row) << 6) + c4;
    async_g2l_b128(gIn,  &sIn [row * 68 + c4]);
    async_g2l_b128(gMem, &sMem[row * 68 + c4]);
    if (tid < 64) sWf[tid] = W_final[tid];
    wait_asynccnt0();
    __syncthreads();

    int mm = tid & 15, ii = tid >> 4;
    const float* pi = &sIn [ii * 68];
    const float* pm = &sMem[mm * 68];
    float s = 0.0f;
    #pragma unroll 8
    for (int d = 0; d < 64; ++d)
        s = fmaf(sWf[d], fast_tanh(pi[d] + pm[d]), s);

    int i = it * 16 + ii;
    int m = mt * 16 + mm;
    float val = (i < ilen[b] && m < mlen[b]) ? s : NEG_INF;
    score[((b * 512 + i) << 10) + m] = val;
}

// ---------------------------------------------------------------------------
// Kernel 3: softmax over Tm=1024 per row; writes attn as bf16
// ---------------------------------------------------------------------------
__global__ void ta_softmax(const float* __restrict__ score, __bf16* __restrict__ attn) {
    __shared__ float red[256];
    int r = blockIdx.x, tid = threadIdx.x;
    const float* srow = score + (r << 10);

    float v0 = srow[tid], v1 = srow[tid + 256], v2 = srow[tid + 512], v3 = srow[tid + 768];
    float mx = fmaxf(fmaxf(v0, v1), fmaxf(v2, v3));
    red[tid] = mx; __syncthreads();
    #pragma unroll
    for (int s = 128; s > 0; s >>= 1) {
        if (tid < s) red[tid] = fmaxf(red[tid], red[tid + s]);
        __syncthreads();
    }
    mx = red[0]; __syncthreads();

    const float L2E = 1.44269504f;
    float e0 = fast_exp2((v0 - mx) * L2E);
    float e1 = fast_exp2((v1 - mx) * L2E);
    float e2 = fast_exp2((v2 - mx) * L2E);
    float e3 = fast_exp2((v3 - mx) * L2E);
    float sum = (e0 + e1) + (e2 + e3);
    red[tid] = sum; __syncthreads();
    #pragma unroll
    for (int s = 128; s > 0; s >>= 1) {
        if (tid < s) red[tid] += red[tid + s];
        __syncthreads();
    }
    float inv = 1.0f / red[0];

    __bf16* arow = attn + (r << 10);
    arow[tid]       = (__bf16)(e0 * inv);
    arow[tid + 256] = (__bf16)(e1 * inv);
    arow[tid + 512] = (__bf16)(e2 * inv);
    arow[tid + 768] = (__bf16)(e3 * inv);
}

// ---------------------------------------------------------------------------
// Kernel 4: out[b,i,d] = sum_m attn[b,i,m] * memory[b,m,d]
// One wave per 16x64 strip of out.  Manually software-pipelined (double
// buffered): iteration kb loads fragments for K-step kb while running the
// WMMAs of K-step kb-32, so loads and WMMAs overlap within the wave.
// K = 1024 -> 32 K-steps, 128 wmma per wave.
// ---------------------------------------------------------------------------
__global__ void ta_outgemm(const __bf16* __restrict__ attn,
                           const __bf16* __restrict__ memT,
                           float* __restrict__ out) {
    int wid  = blockIdx.x * 4 + (threadIdx.x >> 5);
    int lane = threadIdx.x & 31;
    int dg = wid & 3;               // 64-col group (4 of them)
    int it = (wid >> 2) & 31;       // Ti tile  (32 of them)
    int b  = wid >> 7;              // batch    (4)

    int r  = lane & 15;
    int hi = lane >> 4;

    const __bf16* arow = attn + (size_t)(b * 512 + it * 16 + r) * 1024 + hi * 16;
    const __bf16* br0 = memT + (size_t)(b * 256 + dg * 64 +  0 + r) * 1024 + hi * 16;
    const __bf16* br1 = memT + (size_t)(b * 256 + dg * 64 + 16 + r) * 1024 + hi * 16;
    const __bf16* br2 = memT + (size_t)(b * 256 + dg * 64 + 32 + r) * 1024 + hi * 16;
    const __bf16* br3 = memT + (size_t)(b * 256 + dg * 64 + 48 + r) * 1024 + hi * 16;

    v8f c0 = {}, c1 = {}, c2 = {}, c3 = {};

    // Prologue: fragments for K-step 0
    v8bf  alo = *(const v8bf*)(arow);
    v8bf  ahi = *(const v8bf*)(arow + 8);
    v16bf b0  = *(const v16bf*)(br0);
    v16bf b1  = *(const v16bf*)(br1);
    v16bf b2  = *(const v16bf*)(br2);
    v16bf b3  = *(const v16bf*)(br3);

    #pragma unroll 4
    for (int kb = 32; kb < 1024; kb += 32) {
        // Load next K-step's fragments (kept live across the WMMAs below).
        v8bf  aloN = *(const v8bf*)(arow + kb);
        v8bf  ahiN = *(const v8bf*)(arow + kb + 8);
        v16bf b0N  = *(const v16bf*)(br0 + kb);
        v16bf b1N  = *(const v16bf*)(br1 + kb);
        v16bf b2N  = *(const v16bf*)(br2 + kb);
        v16bf b3N  = *(const v16bf*)(br3 + kb);
        __builtin_prefetch(arow + kb + 256, 0, 0);

        // WMMAs on current buffer.
        v16bf a = make_afrag(alo, ahi);
        c0 = wmma_bf16(a, b0, c0);
        c1 = wmma_bf16(a, b1, c1);
        c2 = wmma_bf16(a, b2, c2);
        c3 = wmma_bf16(a, b3, c3);

        alo = aloN; ahi = ahiN;
        b0 = b0N; b1 = b1N; b2 = b2N; b3 = b3N;
    }

    // Epilogue: last K-step
    {
        v16bf a = make_afrag(alo, ahi);
        c0 = wmma_bf16(a, b0, c0);
        c1 = wmma_bf16(a, b1, c1);
        c2 = wmma_bf16(a, b2, c2);
        c3 = wmma_bf16(a, b3, c3);
    }

    float* orow = out + (size_t)(b * 512 + it * 16 + hi * 8) * 256 + dg * 64 + r;
    #pragma unroll
    for (int j = 0; j < 8; ++j) {
        orow[j * 256 +  0] = c0[j];
        orow[j * 256 + 16] = c1[j];
        orow[j * 256 + 32] = c2[j];
        orow[j * 256 + 48] = c3[j];
    }
}

// ---------------------------------------------------------------------------
extern "C" void kernel_launch(void* const* d_in, const int* in_sizes, int n_in,
                              void* d_out, int out_size, void* d_ws, size_t ws_size,
                              hipStream_t stream) {
    const float* inputs  = (const float*)d_in[0];   // (4,512,256)
    const float* memory  = (const float*)d_in[1];   // (4,1024,256)
    const int*   ilen    = (const int*)  d_in[2];   // (4,)
    const int*   mlen    = (const int*)  d_in[3];   // (4,)
    const float* W_in    = (const float*)d_in[4];   // (64,256)
    const float* b_in    = (const float*)d_in[5];   // (64,)
    const float* W_mem   = (const float*)d_in[6];   // (64,256)
    const float* W_final = (const float*)d_in[7];   // (64,)
    float* out = (float*)d_out;                     // (4,512,256)

    char* ws = (char*)d_ws;                          // all offsets 256B-aligned
    __bf16* in_bf    = (__bf16*)(ws + 0);            // 1,048,576 B
    __bf16* mem_bf   = (__bf16*)(ws + 1048576);      // 2,097,152 B
    __bf16* memT     = (__bf16*)(ws + 3145728);      // 2,097,152 B
    __bf16* win_bf   = (__bf16*)(ws + 5242880);      //    32,768 B
    __bf16* wmem_bf  = (__bf16*)(ws + 5275648);      //    32,768 B
    float*  in_item  = (float*) (ws + 5308416);      //   524,288 B (2048x64)
    float*  mem_item = (float*) (ws + 5832704);      // 1,048,576 B (4096x64)
    float*  score    = (float*) (ws + 6881280);      // 8,388,608 B (2048x1024)
    __bf16* attn     = (__bf16*)(ws + 15269888);     // 4,194,304 B (2048x1024)

    // Stage 0: bf16 copies (+ memory transpose for contiguous B-fragments)
    ta_cvt_bf16<<<2048, 256, 0, stream>>>(inputs, in_bf, 524288);
    ta_cvt_mem <<<4096, 256, 0, stream>>>(memory, mem_bf, memT);
    ta_cvt_bf16<<<64,   256, 0, stream>>>(W_in,   win_bf,  16384);
    ta_cvt_bf16<<<64,   256, 0, stream>>>(W_mem,  wmem_bf, 16384);

    // Stage 1: projections via bf16 WMMA (f32 accumulate), 16x64 per wave
    ta_proj<<<32, 128, 0, stream>>>(in_bf,  win_bf,  b_in,    in_item,  128);  // M=2048
    ta_proj<<<64, 128, 0, stream>>>(mem_bf, wmem_bf, nullptr, mem_item, 256);  // M=4096

    // Stage 2: tanh score + mask (async LDS staging)
    dim3 g2(64, 32, 4);
    ta_score<<<g2, 256, 0, stream>>>(in_item, mem_item, W_final, ilen, mlen, score);

    // Stage 3: softmax over Tm, emit bf16 attn
    ta_softmax<<<2048, 256, 0, stream>>>(score, attn);

    // Stage 4: out = attn @ memory via bf16 WMMA, 16x64 per wave
    ta_outgemm<<<128, 128, 0, stream>>>(attn, memT, out);
}
